// MultiHeadVisionAttention_63754494542006
// MI455X (gfx1250) — matrix operand order repaired and checked by reading, not compiled, not run on hardware
//
#include <hip/hip_runtime.h>
#include <hip/hip_bf16.h>

// Problem constants (from reference): B=32, N=1024, H=1024, NH=16, D=64
#define BQ    32
#define NTOK  1024
#define HDIM  1024
#define NHEAD 16
#define DHEAD 64

typedef __attribute__((ext_vector_type(16))) __bf16 bf16x16;
typedef __attribute__((ext_vector_type(8)))  __bf16 bf16x8;
typedef __attribute__((ext_vector_type(8)))  float  f32x8;
typedef __attribute__((ext_vector_type(4)))  unsigned int v4u;
typedef __attribute__((ext_vector_type(8)))  int v8i;
typedef __attribute__((ext_vector_type(4)))  int v4i;

#define WMMA_BF16(a, b, c) \
    __builtin_amdgcn_wmma_f32_16x16x32_bf16(false, (a), false, (b), (short)0, (c), false, false)

#if __has_builtin(__builtin_amdgcn_tensor_load_to_lds)
#define HAVE_TDM 1
#else
#define HAVE_TDM 0
#endif

__device__ __forceinline__ __bf16 f2bf(float f) {
    unsigned u = __builtin_bit_cast(unsigned, f);
    unsigned r = u + 0x7FFFu + ((u >> 16) & 1u);   // round-to-nearest-even
    unsigned short s = (unsigned short)(r >> 16);
    return __builtin_bit_cast(__bf16, s);
}
__device__ __forceinline__ float bf2f(__bf16 b) {
    unsigned u = ((unsigned)__builtin_bit_cast(unsigned short, b)) << 16;
    return __builtin_bit_cast(float, u);
}

// LDS byte offset of a generic pointer into __shared__ (AS3 addresses are offsets
// from the workgroup LDS base, which is what the TDM D# lds_addr field wants).
__device__ __forceinline__ unsigned lds_byte_off(const void* p) {
    return (unsigned)(unsigned long long)(__attribute__((address_space(3))) const void*)p;
}

// A fragment (16x32 bf16, MxK): lane L holds row M=L%16; lanes 0-15 take K={0..7,16..23},
// lanes 16-31 take K={8..15,24..31}. Two 16B contiguous slices per lane.
__device__ __forceinline__ bf16x16 frag_a(const __bf16* base, int stride) {
    const int lane = threadIdx.x & 31;
    const int row  = lane & 15;
    const int kb   = (lane & 16) ? 8 : 0;
    const __bf16* p = base + row * stride + kb;
    bf16x8 lo = *(const bf16x8*)(p);
    bf16x8 hi = *(const bf16x8*)(p + 16);
    return __builtin_shufflevector(lo, hi, 0,1,2,3,4,5,6,7,8,9,10,11,12,13,14,15);
}

// B fragment (32x16 bf16, KxN): lane L holds column N=L%16; lanes 0-15 take K={0..7,16..23},
// lanes 16-31 take K={8..15,24..31}. base is laid out [col][k] row-major with `stride` elems.
__device__ __forceinline__ bf16x16 frag_b(const __bf16* base, int stride) {
    const int lane = threadIdx.x & 31;
    const int col  = lane & 15;
    const int kh   = (lane & 16) ? 8 : 0;
    const __bf16* p = base + col * stride + kh;
    bf16x8 lo = *(const bf16x8*)(p);
    bf16x8 hi = *(const bf16x8*)(p + 16);
    return __builtin_shufflevector(lo, hi, 0,1,2,3,4,5,6,7,8,9,10,11,12,13,14,15);
}

// ---------------------------------------------------------------- K0: fp32 -> bf16
__global__ void cvt_f32_bf16(const float* __restrict__ src, __bf16* __restrict__ dst, int n) {
    int i = blockIdx.x * blockDim.x + threadIdx.x;
    if (i < n) dst[i] = f2bf(src[i]);
}

// ---------------------------------------------------------------- K1: X @ W^T -> head layout
// Outh[(b*16+h)*1024 + n][d] = sum_k X[b*1024+n][k] * W[h*64+d][k]
// Workgroup: 256 thr = 8 waves (2x4), tile 64 rows x 128 cols.
// A tile staged fp32->bf16 via VALU+LDS; B tile (bf16 weights) DMA'd by the Tensor
// Data Mover into LDS with hardware padding (64B rows + 16B pad = 80B stride).
__global__ __launch_bounds__(256)
void proj_head_kernel(const float* __restrict__ X, const __bf16* __restrict__ Wb,
                      __bf16* __restrict__ Outh) {
    __shared__ __align__(16) __bf16 As[64 * 40];
#if HAVE_TDM
    __shared__ __align__(16) __bf16 Bs[128 * 40];
#endif

    const int ct  = blockIdx.x;           // column tile * 128
    const int mt  = blockIdx.y;           // 0..511 (64-token tiles)
    const int b   = mt >> 4;
    const int n0  = (mt & 15) << 6;
    const int rowBase = b * NTOK + n0;

    const int tid  = threadIdx.x;
    const int wave = tid >> 5;
    const int wm   = wave >> 2;           // 0..1
    const int wn   = wave & 3;            // 0..3
    const int colBase = ct * 128 + wn * 32;

    const int lr = tid >> 2;              // 0..63 : A-tile row
    const int lk = (tid & 3) << 3;        // 0,8,16,24 : A-tile k base

#if HAVE_TDM
    // D# group 1 (constant across k-steps): data_size=2B (code 1), pad_enable,
    // pad_interval=3 (pad after every 16 DWORDs = 64B), pad_amount=3 (4 DWORDs = 16B),
    // tensor_dim0 = tile_dim0 = 32 elems, tensor_dim1 = tile_dim1 = 128 rows,
    // tensor_dim0_stride = 1024 elems (weight row pitch).
    v8i g1c;
    g1c[0] = (int)((1u << 16) | (1u << 20) | (3u << 22) | (3u << 25));
    g1c[1] = (int)(32u << 16);      // tensor_dim0[15:0] in bits[31:16]
    g1c[2] = (int)(128u << 16);     // tensor_dim0 hi=0, tensor_dim1[15:0]
    g1c[3] = (int)(32u << 16);      // tensor_dim1 hi=0, tile_dim0=32
    g1c[4] = 128;                   // tile_dim1=128, tile_dim2=0
    g1c[5] = 1024;                  // tensor_dim0_stride lo
    g1c[6] = 0;                     // stride0 hi, stride1 lo
    g1c[7] = 0;                     // stride1 hi
    const v4i z4 = {0, 0, 0, 0};
    const unsigned bs_off = lds_byte_off(&Bs[0]);
#endif

    f32x8 acc[2][2] = {};

    for (int kk = 0; kk < HDIM; kk += 32) {
        __syncthreads();
#if HAVE_TDM
        if (wave == 0) {
            const unsigned long long ga =
                (unsigned long long)(const void*)(Wb + (size_t)ct * 128 * HDIM + kk);
            v4u g0;
            g0[0] = 1u;                                      // count=1, user descriptor
            g0[1] = bs_off;                                  // lds_addr
            g0[2] = (unsigned)ga;                            // global_addr[95:64]
            g0[3] = (unsigned)((ga >> 32) & 0x01FFFFFFull)   // global_addr[120:96]
                    | 0x80000000u;                           // type=2 ("image")
#if defined(__clang_major__) && __clang_major__ >= 23
            v8i g1pad = {0, 0, 0, 0, 0, 0, 0, 0};
            __builtin_amdgcn_tensor_load_to_lds(g0, g1c, z4, z4, g1pad, 0);
#else
            __builtin_amdgcn_tensor_load_to_lds(g0, g1c, z4, z4, 0);
#endif
        }
#endif
        // A tile 64x32 fp32 -> bf16 in LDS (one ds_store_b128 per thread)
        const float* src = X + (size_t)(rowBase + lr) * HDIM + kk + lk;
        float4 v0 = *(const float4*)(src);
        float4 v1 = *(const float4*)(src + 4);
        bf16x8 pk;
        pk[0] = f2bf(v0.x); pk[1] = f2bf(v0.y); pk[2] = f2bf(v0.z); pk[3] = f2bf(v0.w);
        pk[4] = f2bf(v1.x); pk[5] = f2bf(v1.y); pk[6] = f2bf(v1.z); pk[7] = f2bf(v1.w);
        *(bf16x8*)(&As[lr * 40 + lk]) = pk;

#if HAVE_TDM
        if (wave == 0) {
#if __has_builtin(__builtin_amdgcn_s_wait_tensorcnt)
            __builtin_amdgcn_s_wait_tensorcnt(0);
#else
            asm volatile("s_wait_tensorcnt 0x0" ::: "memory");
#endif
        }
#endif
        __syncthreads();

        bf16x16 a0 = frag_a(&As[(wm * 32 + 0)  * 40], 40);
        bf16x16 a1 = frag_a(&As[(wm * 32 + 16) * 40], 40);
#if HAVE_TDM
        bf16x16 b0 = frag_b(&Bs[(wn * 32 + 0)  * 40], 40);
        bf16x16 b1 = frag_b(&Bs[(wn * 32 + 16) * 40], 40);
#else
        const __bf16* wp = Wb + (size_t)colBase * HDIM + kk;
        bf16x16 b0 = frag_b(wp, HDIM);
        bf16x16 b1 = frag_b(wp + (size_t)16 * HDIM, HDIM);
#endif

        acc[0][0] = WMMA_BF16(a0, b0, acc[0][0]);
        acc[0][1] = WMMA_BF16(a0, b1, acc[0][1]);
        acc[1][0] = WMMA_BF16(a1, b0, acc[1][0]);
        acc[1][1] = WMMA_BF16(a1, b1, acc[1][1]);
    }

    const int lane = tid & 31;
    const int erow = (lane & 16) ? 8 : 0;
    const int ncol = lane & 15;
#pragma unroll
    for (int mi = 0; mi < 2; mi++)
#pragma unroll
        for (int ni = 0; ni < 2; ni++) {
            const int cglob = colBase + ni * 16 + ncol;
            const int h = cglob >> 6, d = cglob & 63;
#pragma unroll
            for (int e = 0; e < 8; e++) {
                const int n = n0 + wm * 32 + mi * 16 + e + erow;
                Outh[((size_t)(b * NHEAD + h) * NTOK + n) * DHEAD + d] = f2bf(acc[mi][ni][e]);
            }
        }
}

// ---------------------------------------------------------------- K2: per-head attention
// S = q^T (k+pe) / 8 ; P = softmax_rows(S) ; O = V @ P^T, written back to [B,N,H] layout
__global__ __launch_bounds__(128)
void attn_kernel(const __bf16* __restrict__ Qh, const __bf16* __restrict__ Kh,
                 const __bf16* __restrict__ Vh, const __bf16* __restrict__ PEb,
                 __bf16* __restrict__ AOut) {
    __shared__ __align__(16) __bf16 Qt[64 * 40];
    __shared__ __align__(16) __bf16 Kt[64 * 40];
    __shared__ __align__(16) float  Sf[64 * 68];
    __shared__ __align__(16) __bf16 Pb[64 * 72];

    const int b2   = blockIdx.x;               // 0..511 head-batch
    const int tid  = threadIdx.x;
    const int wave = tid >> 5;
    const int lane = tid & 31;
    const size_t hb = (size_t)b2 * NTOK * DHEAD;

    // ---- phase 1: scores (M=64 d, N=64 e, K=1024 n), wave grid 2x2, 32x32 each
    const int sm = wave >> 1, sn = wave & 1;
    const int ln = tid >> 2;                   // 0..31 local token
    const int ld = (tid & 3) << 4;             // 0,16,32,48 feature base
    f32x8 s[2][2] = {};

    for (int nn = 0; nn < NTOK; nn += 32) {
        __syncthreads();
        const __bf16* qp = Qh + hb + (size_t)(nn + ln) * DHEAD + ld;
        const __bf16* kp = Kh + hb + (size_t)(nn + ln) * DHEAD + ld;
        const __bf16* pp = PEb + (size_t)(nn + ln) * DHEAD + ld;
        bf16x8 q0 = *(const bf16x8*)(qp), q1 = *(const bf16x8*)(qp + 8);
        bf16x8 k0 = *(const bf16x8*)(kp), k1 = *(const bf16x8*)(kp + 8);
        bf16x8 p0 = *(const bf16x8*)(pp), p1 = *(const bf16x8*)(pp + 8);
#pragma unroll
        for (int j = 0; j < 8; j++) {          // transpose into [feature][token]
            Qt[(ld + j)     * 40 + ln] = q0[j];
            Qt[(ld + 8 + j) * 40 + ln] = q1[j];
            Kt[(ld + j)     * 40 + ln] = f2bf(bf2f(k0[j]) + bf2f(p0[j]));
            Kt[(ld + 8 + j) * 40 + ln] = f2bf(bf2f(k1[j]) + bf2f(p1[j]));
        }
        __syncthreads();

        bf16x16 a0 = frag_a(&Qt[(sm * 32 + 0)  * 40], 40);
        bf16x16 a1 = frag_a(&Qt[(sm * 32 + 16) * 40], 40);
        bf16x16 b0 = frag_b(&Kt[(sn * 32 + 0)  * 40], 40);
        bf16x16 b1 = frag_b(&Kt[(sn * 32 + 16) * 40], 40);
        s[0][0] = WMMA_BF16(a0, b0, s[0][0]);
        s[0][1] = WMMA_BF16(a0, b1, s[0][1]);
        s[1][0] = WMMA_BF16(a1, b0, s[1][0]);
        s[1][1] = WMMA_BF16(a1, b1, s[1][1]);
    }

    __syncthreads();
    {
        const int erow = (lane & 16) ? 8 : 0;
        const int ncol = lane & 15;
#pragma unroll
        for (int mi = 0; mi < 2; mi++)
#pragma unroll
            for (int ni = 0; ni < 2; ni++)
#pragma unroll
                for (int e = 0; e < 8; e++)
                    Sf[(sm * 32 + mi * 16 + e + erow) * 68 + sn * 32 + ni * 16 + ncol] =
                        s[mi][ni][e] * 0.125f;   // 1/sqrt(64)
    }
    __syncthreads();

    if (tid < 64) {                             // one row of softmax per thread
        float row[64];
        float m = -1e30f;
#pragma unroll
        for (int e = 0; e < 64; e++) { row[e] = Sf[tid * 68 + e]; m = fmaxf(m, row[e]); }
        float sum = 0.f;
#pragma unroll
        for (int e = 0; e < 64; e++) { row[e] = __expf(row[e] - m); sum += row[e]; }
        const float inv = 1.f / sum;
#pragma unroll
        for (int e = 0; e < 64; e++) Pb[tid * 72 + e] = f2bf(row[e] * inv);
    }
    __syncthreads();

    // ---- phase 2: O[n][d] = sum_e P[d][e] * V[n][e]; P fragments stay resident
    bf16x16 bp[2][4];
#pragma unroll
    for (int ks = 0; ks < 2; ks++)
#pragma unroll
        for (int ni = 0; ni < 4; ni++)
            bp[ks][ni] = frag_b(&Pb[(ni * 16) * 72 + ks * 32], 72);

    const int b = b2 >> 4, h = b2 & 15;
    const int erow = (lane & 16) ? 8 : 0;
    const int ncol = lane & 15;

    for (int it = 0; it < 8; it++) {
        const int t0 = it * 128 + wave * 32;
        f32x8 o[2][4] = {};
#pragma unroll
        for (int ks = 0; ks < 2; ks++) {
            bf16x16 a0 = frag_a(Vh + hb + (size_t)t0 * DHEAD + ks * 32, DHEAD);
            bf16x16 a1 = frag_a(Vh + hb + (size_t)(t0 + 16) * DHEAD + ks * 32, DHEAD);
#pragma unroll
            for (int ni = 0; ni < 4; ni++) {
                o[0][ni] = WMMA_BF16(a0, bp[ks][ni], o[0][ni]);
                o[1][ni] = WMMA_BF16(a1, bp[ks][ni], o[1][ni]);
            }
        }
#pragma unroll
        for (int mi = 0; mi < 2; mi++)
#pragma unroll
            for (int ni = 0; ni < 4; ni++) {
                const int d = ni * 16 + ncol;
#pragma unroll
                for (int e = 0; e < 8; e++) {
                    const int n = t0 + mi * 16 + e + erow;
                    AOut[((size_t)(b * NTOK + n)) * HDIM + h * DHEAD + d] = f2bf(o[mi][ni][e]);
                }
            }
    }
}

// ---------------------------------------------------------------- K3: AOut @ Wo^T -> fp32 d_out
__global__ __launch_bounds__(256)
void out_proj_kernel(const __bf16* __restrict__ A, const __bf16* __restrict__ Wo,
                     float* __restrict__ C) {
    const int ct   = blockIdx.x;
    const int mt   = blockIdx.y;
    const int tid  = threadIdx.x;
    const int wave = tid >> 5;
    const int wm = wave >> 2, wn = wave & 3;
    const int rowBase = mt * 64 + wm * 32;
    const int colBase = ct * 128 + wn * 32;

    f32x8 acc[2][2] = {};
    for (int kk = 0; kk < HDIM; kk += 32) {
        __builtin_prefetch(A + (size_t)rowBase * HDIM + kk + 128, 0, 0);
        bf16x16 a0 = frag_a(A + (size_t)rowBase * HDIM + kk, HDIM);
        bf16x16 a1 = frag_a(A + (size_t)(rowBase + 16) * HDIM + kk, HDIM);
        bf16x16 b0 = frag_b(Wo + (size_t)colBase * HDIM + kk, HDIM);
        bf16x16 b1 = frag_b(Wo + (size_t)(colBase + 16) * HDIM + kk, HDIM);
        acc[0][0] = WMMA_BF16(a0, b0, acc[0][0]);
        acc[0][1] = WMMA_BF16(a0, b1, acc[0][1]);
        acc[1][0] = WMMA_BF16(a1, b0, acc[1][0]);
        acc[1][1] = WMMA_BF16(a1, b1, acc[1][1]);
    }

    const int lane = tid & 31;
    const int erow = (lane & 16) ? 8 : 0;
    const int ncol = lane & 15;
#pragma unroll
    for (int mi = 0; mi < 2; mi++)
#pragma unroll
        for (int ni = 0; ni < 2; ni++) {
            float* cp = C + (size_t)(rowBase + mi * 16 + erow) * HDIM + colBase + ni * 16 + ncol;
#pragma unroll
            for (int e = 0; e < 8; e++) cp[(size_t)e * HDIM] = acc[mi][ni][e];
        }
}

// ---------------------------------------------------------------- launch
extern "C" void kernel_launch(void* const* d_in, const int* in_sizes, int n_in,
                              void* d_out, int out_size, void* d_ws, size_t ws_size,
                              hipStream_t stream) {
    const float* query  = (const float*)d_in[0];
    const float* key    = (const float*)d_in[1];
    const float* value  = (const float*)d_in[2];
    const float* key_pe = (const float*)d_in[3];
    const float* Wq = (const float*)d_in[4];
    const float* Wk = (const float*)d_in[5];
    const float* Wv = (const float*)d_in[6];
    const float* Wo = (const float*)d_in[7];

    char*  ws  = (char*)d_ws;
    size_t off = 0;
    auto alloc = [&](size_t bytes) -> void* {
        void* p = ws + off;
        off += (bytes + 255) & ~(size_t)255;
        return p;
    };
    const size_t WELEMS = (size_t)HDIM * HDIM;                 // 1M
    __bf16* Wq_b = (__bf16*)alloc(WELEMS * 2);
    __bf16* Wk_b = (__bf16*)alloc(WELEMS * 2);
    __bf16* Wv_b = (__bf16*)alloc(WELEMS * 2);
    __bf16* Wo_b = (__bf16*)alloc(WELEMS * 2);
    __bf16* PE_b = (__bf16*)alloc((size_t)NTOK * DHEAD * 2);
    const size_t HEADELEMS = (size_t)BQ * NHEAD * NTOK * DHEAD; // 32M
    __bf16* Qh   = (__bf16*)alloc(HEADELEMS * 2);
    __bf16* Kh   = (__bf16*)alloc(HEADELEMS * 2);
    __bf16* Vh   = (__bf16*)alloc(HEADELEMS * 2);
    __bf16* AOut = (__bf16*)alloc((size_t)BQ * NTOK * HDIM * 2);

    cvt_f32_bf16<<<(int)(WELEMS / 256), 256, 0, stream>>>(Wq, Wq_b, (int)WELEMS);
    cvt_f32_bf16<<<(int)(WELEMS / 256), 256, 0, stream>>>(Wk, Wk_b, (int)WELEMS);
    cvt_f32_bf16<<<(int)(WELEMS / 256), 256, 0, stream>>>(Wv, Wv_b, (int)WELEMS);
    cvt_f32_bf16<<<(int)(WELEMS / 256), 256, 0, stream>>>(Wo, Wo_b, (int)WELEMS);
    cvt_f32_bf16<<<NTOK * DHEAD / 256, 256, 0, stream>>>(key_pe, PE_b, NTOK * DHEAD);

    dim3 g1(8, 512);
    proj_head_kernel<<<g1, 256, 0, stream>>>(query, Wq_b, Qh);
    proj_head_kernel<<<g1, 256, 0, stream>>>(key,   Wk_b, Kh);
    proj_head_kernel<<<g1, 256, 0, stream>>>(value, Wv_b, Vh);

    attn_kernel<<<BQ * NHEAD, 128, 0, stream>>>(Qh, Kh, Vh, PE_b, AOut);

    out_proj_kernel<<<g1, 256, 0, stream>>>(AOut, Wo_b, (float*)d_out);
}
